// TF_33578054320567
// MI455X (gfx1250) — matrix operand order repaired and checked by reading, not compiled, not run on hardware
//
#include <hip/hip_runtime.h>
#include <math.h>

// ---------------------------------------------------------------------------
// Problem constants (match the reference)
// ---------------------------------------------------------------------------
#define N_NODES  50000
#define N_EDGES  800000
#define N_GRAPHS 2000
#define IN_CH    64
#define HEADS    8
#define CHAN     16
#define EDGE_D   16
#define HIDW     128            // HEADS*CHAN
#define CLSW     512
#define CATW     (HIDW * 5)     // 640 = HID*(L+2)
#define SCALE_A  0.25f          // 1/sqrt(CHAN)

#define BM 128
#define BN 64
#define BK 32
#define LDSP 40                 // LDS pitch in bf16 (80B = 5*16B: aligned + bank-spread)

typedef __attribute__((ext_vector_type(16))) __bf16 bf16x16;
typedef __attribute__((ext_vector_type(8)))  __bf16 bf16x8;
typedef __attribute__((ext_vector_type(4)))  __bf16 bf16x4;
typedef __attribute__((ext_vector_type(8)))  float  f32x8;

// ---------------------------------------------------------------------------
// Tiled GEMM:  C[M,N] = act( A[M,K] @ B[K,N] + bias[N] )
// fp32 in/out, bf16 WMMA with fp32 accumulate (v_wmma_f32_16x16x32_bf16).
// Block: 128(M) x 64(N), 128 threads = 4 waves. Wave wv owns rows
// [bm+wv*32, +32) as 2 M-tiles and all 4 N-tiles: 8 WMMAs per K-step versus
// 12 ds_load_b128 (B fragments shared across the 2 M-tiles).
// Fragment layout per ISA 7.12.2:
//   A 16x32 bf16: lane=(kgrp<<4)|row -> A[row, kb..kb+7], A[row, kb+16..kb+23], kb=kgrp*8
//   B 32x16 bf16: lane=(kgrp<<4)|col -> B[kb..kb+7, col], B[kb+16..kb+23, col] (LDS holds B transposed)
// No guarded slow path: out-of-range M rows / N cols are CLAMPED during
// staging (garbage flows only into accumulator rows/cols the epilogue never
// stores), keeping control flow uniform and loads fully vectorized. The
// K-tail (K=16 edge GEMM) is a per-float4/per-pair zero-fill select since it
// feeds every accumulation.
// ---------------------------------------------------------------------------
__global__ __launch_bounds__(128)
void gemm_bf16_wmma(const float* __restrict__ A, const float* __restrict__ B,
                    const float* __restrict__ bias, float* __restrict__ C,
                    int M, int N, int K, int act /*0 none, 1 leaky-relu*/) {
  __shared__ __bf16 As[BM * LDSP];   // As[r*40 + k]
  __shared__ __bf16 Bs[BN * LDSP];   // Bs[n*40 + k]  (transposed)

  const int tid   = threadIdx.x;
  const int lane  = tid & 31;
  const int wv    = tid >> 5;
  const int bm    = blockIdx.y * BM;
  const int bn    = blockIdx.x * BN;
  const int row16 = lane & 15;
  const int kgrp  = lane >> 4;

  f32x8 acc[2][4] = {};

  for (int k0 = 0; k0 < K; k0 += BK) {
    // ---- A tile 128x32: 8 float4 loads/thread (rows clamped), bf16x4 stores ----
#pragma unroll
    for (int i = 0; i < 8; ++i) {
      int f = i * 128 + tid;            // float4 slot 0..1023
      int row = f >> 3, c4 = f & 7;
      int gr = min(bm + row, M - 1);    // clamp: epilogue masks these rows
      int gc = k0 + c4 * 4;
      float4 av = make_float4(0.f, 0.f, 0.f, 0.f);
      if (gc < K) av = *(const float4*)&A[(size_t)gr * K + gc];
      *(bf16x4*)&As[row * LDSP + c4 * 4] =
          (bf16x4){(__bf16)av.x, (__bf16)av.y, (__bf16)av.z, (__bf16)av.w};
    }
    // ---- B tile 32x64 transposed: 16 coalesced loads/thread, 2 b128 stores ----
    {
      int n = tid & 63, khalf = tid >> 6;   // khalf: 0/1 -> k 0..15 / 16..31
      int gn = min(bn + n, N - 1);          // clamp: epilogue masks these cols
      unsigned int pk[8];
#pragma unroll
      for (int j = 0; j < 8; ++j) {
        int gk = k0 + khalf * 16 + 2 * j;
        float b0 = (gk     < K) ? B[(size_t)gk * N + gn]       : 0.f;
        float b1 = (gk + 1 < K) ? B[(size_t)(gk + 1) * N + gn] : 0.f;
        union { unsigned int u; __bf16 h[2]; } p;
        p.h[0] = (__bf16)b0; p.h[1] = (__bf16)b1;
        pk[j] = p.u;
      }
      uint4* bd = (uint4*)&Bs[n * LDSP + khalf * 16];
      bd[0] = make_uint4(pk[0], pk[1], pk[2], pk[3]);
      bd[1] = make_uint4(pk[4], pk[5], pk[6], pk[7]);
    }
    // prefetch next A K-slab (global_prefetch_b8)
    if (k0 + BK < K) {
      int pr = min(bm + tid, M - 1);
      __builtin_prefetch(&A[(size_t)pr * K + k0 + BK], 0, 1);
    }
    __syncthreads();

    // ---- A fragments: 2 M-tiles per wave ----
    bf16x16 afrag[2];
#pragma unroll
    for (int mt = 0; mt < 2; ++mt) {
      const __bf16* ap = &As[(wv * 32 + mt * 16 + row16) * LDSP + kgrp * 8];
      bf16x8 lo = *(const bf16x8*)ap;          // K = kb+0..7
      bf16x8 hi = *(const bf16x8*)(ap + 16);   // K = kb+16..23
#pragma unroll
      for (int i = 0; i < 8; ++i) { afrag[mt][i] = lo[i]; afrag[mt][8 + i] = hi[i]; }
    }
    // ---- 4 N-tiles x 2 M-tiles = 8 WMMAs ----
#pragma unroll
    for (int nt = 0; nt < 4; ++nt) {
      const __bf16* bp = &Bs[(nt * 16 + row16) * LDSP + kgrp * 8];
      bf16x8 blo = *(const bf16x8*)bp;
      bf16x8 bhi = *(const bf16x8*)(bp + 16);
      bf16x16 bfrag;
#pragma unroll
      for (int i = 0; i < 8; ++i) { bfrag[i] = blo[i]; bfrag[8 + i] = bhi[i]; }
      acc[0][nt] = __builtin_amdgcn_wmma_f32_16x16x32_bf16(
          false, afrag[0], false, bfrag, (short)0, acc[0][nt], false, false);
      acc[1][nt] = __builtin_amdgcn_wmma_f32_16x16x32_bf16(
          false, afrag[1], false, bfrag, (short)0, acc[1][nt], false, false);
    }
    __syncthreads();
  }

  // ---- epilogue: D layout (lane<16: M=r, lane>=16: M=r+8), col = lane&15 ----
#pragma unroll
  for (int mt = 0; mt < 2; ++mt) {
#pragma unroll
    for (int nt = 0; nt < 4; ++nt) {
      int col = bn + nt * 16 + row16;
      if (col >= N) continue;
      float bb = bias ? bias[col] : 0.0f;
#pragma unroll
      for (int r = 0; r < 8; ++r) {
        int grow = bm + wv * 32 + mt * 16 + r + kgrp * 8;
        if (grow < M) {
          float v = acc[mt][nt][r] + bb;
          if (act == 1) v = (v > 0.0f) ? v : 0.01f * v;
          C[(size_t)grow * N + col] = v;
        }
      }
    }
  }
}

// ---------------------------------------------------------------------------
// Edge-wise attention kernels
// ---------------------------------------------------------------------------
__device__ __forceinline__ void atomicMaxF(float* addr, float val) {
  // ordered-int trick: correct for mixed signs, init must be -FLT_MAX
  if (val >= 0.0f) atomicMax((int*)addr, __float_as_int(val));
  else             atomicMin((unsigned int*)addr, __float_as_uint(val));
}

__global__ __launch_bounds__(256)
void init_seg(float* __restrict__ agg, float* __restrict__ amax,
              float* __restrict__ denom, int nhid, int nh) {
  int i = blockIdx.x * blockDim.x + threadIdx.x;
  if (i < nhid) agg[i] = 0.0f;
  if (i < nh) { amax[i] = -3.402823466e38f; denom[i] = 0.0f; }
}

// pass 1: alpha[e,h] = SCALE * dot(q[dst,h,:], k[src,h,:] + e[e,h,:]); seg-max
__global__ __launch_bounds__(256)
void attn_alpha(const int* __restrict__ ei, const float* __restrict__ q,
                const float* __restrict__ k, const float* __restrict__ ebuf,
                float* __restrict__ alpha, float* __restrict__ amax) {
  int t = blockIdx.x * blockDim.x + threadIdx.x;
  if (t >= N_EDGES * HEADS) return;
  int e = t >> 3, h = t & 7;
  int src = ei[e], dst = ei[N_EDGES + e];
  const float4* qp = (const float4*)&q[(size_t)dst * HIDW + h * CHAN];
  const float4* kp = (const float4*)&k[(size_t)src * HIDW + h * CHAN];
  const float4* ep = (const float4*)&ebuf[(size_t)e * HIDW + h * CHAN];
  float s = 0.0f;
#pragma unroll
  for (int i = 0; i < 4; ++i) {
    float4 qv = qp[i], kv = kp[i], ev = ep[i];
    s += qv.x * (kv.x + ev.x) + qv.y * (kv.y + ev.y) +
         qv.z * (kv.z + ev.z) + qv.w * (kv.w + ev.w);
  }
  s *= SCALE_A;
  alpha[t] = s;
  atomicMaxF(&amax[dst * HEADS + h], s);
}

// pass 2: ex = exp(alpha - amax[dst]); seg-sum
__global__ __launch_bounds__(256)
void attn_expsum(const int* __restrict__ ei, float* __restrict__ alpha,
                 const float* __restrict__ amax, float* __restrict__ denom) {
  int t = blockIdx.x * blockDim.x + threadIdx.x;
  if (t >= N_EDGES * HEADS) return;
  int e = t >> 3, h = t & 7;
  int dst = ei[N_EDGES + e];
  float ex = __expf(alpha[t] - amax[dst * HEADS + h]);
  alpha[t] = ex;
  atomicAdd(&denom[dst * HEADS + h], ex);
}

// pass 3: agg[dst] += (v[src]+e) * ex/denom[dst]
__global__ __launch_bounds__(256)
void attn_msg(const int* __restrict__ ei, const float* __restrict__ v,
              const float* __restrict__ ebuf, const float* __restrict__ alpha,
              const float* __restrict__ denom, float* __restrict__ agg) {
  int t = blockIdx.x * blockDim.x + threadIdx.x;
  if (t >= N_EDGES * HEADS) return;
  int e = t >> 3, h = t & 7;
  int src = ei[e], dst = ei[N_EDGES + e];
  float attn = alpha[t] / denom[dst * HEADS + h];
  const float* vp = &v[(size_t)src * HIDW + h * CHAN];
  const float* ep = &ebuf[(size_t)e * HIDW + h * CHAN];
  float* ap = &agg[(size_t)dst * HIDW + h * CHAN];
#pragma unroll
  for (int c = 0; c < CHAN; ++c) atomicAdd(&ap[c], (vp[c] + ep[c]) * attn);
}

__global__ __launch_bounds__(256)
void add_skip_lrelu(const float* __restrict__ agg, const float* __restrict__ skip,
                    float* __restrict__ out, int n) {
  int i = blockIdx.x * blockDim.x + threadIdx.x;
  if (i < n) {
    float v = agg[i] + skip[i];
    out[i] = (v > 0.0f) ? v : 0.01f * v;
  }
}

// ---------------------------------------------------------------------------
// Pool / concat / final head
// ---------------------------------------------------------------------------
__global__ __launch_bounds__(256)
void fill_zero(float* __restrict__ p, int n) {
  int i = blockIdx.x * blockDim.x + threadIdx.x;
  if (i < n) p[i] = 0.0f;
}

__global__ __launch_bounds__(256)
void pool_add(const float* __restrict__ h, const int* __restrict__ batch,
              float* __restrict__ pool) {
  int i = blockIdx.x * blockDim.x + threadIdx.x;
  if (i >= N_NODES * HIDW) return;
  int n = i >> 7, j = i & 127;
  atomicAdd(&pool[(size_t)batch[n] * HIDW + j], h[i]);
}

__global__ __launch_bounds__(256)
void concat_k(const float* __restrict__ h0, const float* __restrict__ h1,
              const float* __restrict__ h2, const float* __restrict__ h3,
              const float* __restrict__ pool, const int* __restrict__ batch,
              float* __restrict__ cat) {
  int i = blockIdx.x * blockDim.x + threadIdx.x;
  if (i >= N_NODES * CATW) return;
  int n = i / CATW;
  int j = i - n * CATW;
  float v;
  if      (j < 128) v = h0[(size_t)n * HIDW + j];
  else if (j < 256) v = h1[(size_t)n * HIDW + (j - 128)];
  else if (j < 384) v = h2[(size_t)n * HIDW + (j - 256)];
  else if (j < 512) v = h3[(size_t)n * HIDW + (j - 384)];
  else              v = pool[(size_t)batch[n] * HIDW + (j - 512)];
  cat[(size_t)n * CATW + j] = v;
}

// one wave per node: out = sigmoid(h @ W_fin + b_fin)
__global__ __launch_bounds__(256)
void final_k(const float* __restrict__ h, const float* __restrict__ W,
             const float* __restrict__ b, float* __restrict__ out) {
  int gid  = blockIdx.x * blockDim.x + threadIdx.x;
  int node = gid >> 5;
  int lane = gid & 31;
  if (node >= N_NODES) return;
  float s = 0.0f;
  for (int kk = lane; kk < CLSW; kk += 32)
    s += h[(size_t)node * CLSW + kk] * W[kk];
#pragma unroll
  for (int o = 16; o > 0; o >>= 1) s += __shfl_xor(s, o, 32);
  if (lane == 0) out[node] = 1.0f / (1.0f + __expf(-(s + b[0])));
}

// ---------------------------------------------------------------------------
// Host-side orchestration
// ---------------------------------------------------------------------------
extern "C" void kernel_launch(void* const* d_in, const int* in_sizes, int n_in,
                              void* d_out, int out_size, void* d_ws, size_t ws_size,
                              hipStream_t stream) {
  const float* x     = (const float*)d_in[0];
  const int*   ei    = (const int*)  d_in[1];
  const float* eattr = (const float*)d_in[2];
  const int*   batch = (const int*)  d_in[3];
  const float* Wq1 = (const float*)d_in[4];  const float* bq1 = (const float*)d_in[5];
  const float* Wk1 = (const float*)d_in[6];  const float* bk1 = (const float*)d_in[7];
  const float* Wv1 = (const float*)d_in[8];  const float* bv1 = (const float*)d_in[9];
  const float* We1 = (const float*)d_in[10];
  const float* Ws1 = (const float*)d_in[11]; const float* bs1 = (const float*)d_in[12];
  const float* cWq = (const float*)d_in[13]; const float* cbq = (const float*)d_in[14];
  const float* cWk = (const float*)d_in[15]; const float* cbk = (const float*)d_in[16];
  const float* cWv = (const float*)d_in[17]; const float* cbv = (const float*)d_in[18];
  const float* cWe = (const float*)d_in[19];
  const float* cWs = (const float*)d_in[20]; const float* cbs = (const float*)d_in[21];
  const float* Wc1 = (const float*)d_in[22]; const float* bc1 = (const float*)d_in[23];
  const float* clsW = (const float*)d_in[24]; const float* clsB = (const float*)d_in[25];
  const float* Wfin = (const float*)d_in[26]; const float* bfin = (const float*)d_in[27];

  // ---- workspace carve-up (floats). Peak ~670 MB. ----
  float* ws = (float*)d_ws;
  size_t off = 0;
  float* hs[4];
  for (int i = 0; i < 4; ++i) { hs[i] = ws + off; off += (size_t)N_NODES * HIDW; }
  float* qb = ws + off; off += (size_t)N_NODES * HIDW;
  float* kb = ws + off; off += (size_t)N_NODES * HIDW;
  float* vb = ws + off; off += (size_t)N_NODES * HIDW;
  float* sb = ws + off; off += (size_t)N_NODES * HIDW;
  float* ebuf = ws + off; off += (size_t)N_EDGES * HIDW;       // 409 MB, reused below
  float* alph = ws + off; off += (size_t)N_EDGES * HEADS;
  float* amax = ws + off; off += (size_t)N_NODES * HEADS;
  float* denom = ws + off; off += (size_t)N_NODES * HEADS;
  float* agg  = ws + off; off += (size_t)N_NODES * HIDW;
  float* pool = ws + off; off += (size_t)N_GRAPHS * HIDW;
  // classifier buffers alias ebuf (edge embeddings dead by then): 83.2M <= 102.4M floats
  float* cat = ebuf;
  float* c1  = ebuf + (size_t)N_NODES * CATW;
  float* c2  = c1   + (size_t)N_NODES * CLSW;

  auto gemm = [&](const float* A, const float* B, const float* bias, float* C,
                  int M, int N, int K, int act) {
    dim3 g((N + BN - 1) / BN, (M + BM - 1) / BM);
    gemm_bf16_wmma<<<g, 128, 0, stream>>>(A, B, bias, C, M, N, K, act);
  };
  const int EH = N_EDGES * HEADS;
  const int NH = N_NODES * HIDW;

  const float* in = x;
  int kin = IN_CH;
  for (int l = 0; l < 4; ++l) {
    const float *Wq, *bq, *Wk, *bk, *Wv, *bv, *We, *Wsk, *bsk;
    if (l == 0) {
      Wq = Wq1; bq = bq1; Wk = Wk1; bk = bk1; Wv = Wv1; bv = bv1;
      We = We1; Wsk = Ws1; bsk = bs1;
    } else {
      int i = l - 1;
      Wq = cWq + (size_t)i * HIDW * HIDW; bq = cbq + (size_t)i * HIDW;
      Wk = cWk + (size_t)i * HIDW * HIDW; bk = cbk + (size_t)i * HIDW;
      Wv = cWv + (size_t)i * HIDW * HIDW; bv = cbv + (size_t)i * HIDW;
      We = cWe + (size_t)i * EDGE_D * HIDW;
      Wsk = cWs + (size_t)i * HIDW * HIDW; bsk = cbs + (size_t)i * HIDW;
    }
    // node projections + skip (WMMA GEMMs)
    gemm(in, Wq, bq, qb, N_NODES, HIDW, kin, 0);
    gemm(in, Wk, bk, kb, N_NODES, HIDW, kin, 0);
    gemm(in, Wv, bv, vb, N_NODES, HIDW, kin, 0);
    gemm(in, Wsk, bsk, sb, N_NODES, HIDW, kin, 0);
    // edge embedding GEMM (K=16 zero-padded to 32 inside the kernel)
    gemm(eattr, We, nullptr, ebuf, N_EDGES, HIDW, EDGE_D, 0);
    // segment softmax + scatter
    init_seg<<<(NH + 255) / 256, 256, 0, stream>>>(agg, amax, denom, NH, N_NODES * HEADS);
    attn_alpha<<<(EH + 255) / 256, 256, 0, stream>>>(ei, qb, kb, ebuf, alph, amax);
    attn_expsum<<<(EH + 255) / 256, 256, 0, stream>>>(ei, alph, amax, denom);
    attn_msg<<<(EH + 255) / 256, 256, 0, stream>>>(ei, vb, ebuf, alph, denom, agg);
    add_skip_lrelu<<<(NH + 255) / 256, 256, 0, stream>>>(agg, sb, hs[l], NH);
    in = hs[l];
    kin = HIDW;
  }

  // global add pool over the last layer, broadcast back per node
  fill_zero<<<(N_GRAPHS * HIDW + 255) / 256, 256, 0, stream>>>(pool, N_GRAPHS * HIDW);
  pool_add<<<(NH + 255) / 256, 256, 0, stream>>>(hs[3], batch, pool);
  concat_k<<<(int)(((size_t)N_NODES * CATW + 255) / 256), 256, 0, stream>>>(
      hs[0], hs[1], hs[2], hs[3], pool, batch, cat);

  // classifier head (WMMA GEMMs)
  gemm(cat, Wc1, bc1, c1, N_NODES, CLSW, CATW, 0);                       // no activation
  gemm(c1, clsW, clsB, c2, N_NODES, CLSW, CLSW, 1);                      // lrelu
  gemm(c2, clsW + (size_t)CLSW * CLSW, clsB + CLSW, c1, N_NODES, CLSW, CLSW, 1);
  final_k<<<(int)(((size_t)N_NODES * 32 + 255) / 256), 256, 0, stream>>>(
      c1, Wfin, bfin, (float*)d_out);
}
